// PointsDecoder_13675175871191
// MI455X (gfx1250) — matrix operand order repaired and checked by reading, not compiled, not run on hardware
//
#include <hip/hip_runtime.h>
#include <hip/hip_bf16.h>
#include <math.h>

#define NB   2
#define MM   16384
#define PP   5023
#define KK   4
#define CPL  32
#define HPD  256

typedef __attribute__((ext_vector_type(16))) _Float16 v16h;
typedef __attribute__((ext_vector_type(8)))  _Float16 v8h;
typedef __attribute__((ext_vector_type(8)))  float    v8f;

union AF { v16h v; v8h h[2]; };

// Transposed/padded f16 weight offsets (in halfs) inside the ws weight arena.
#define OW_PW1 0        // [64 cols][64 Kpad]   (rows 59 -> 64)
#define OW_PW2 4096     // [32][64]
#define OW_FW1 6144     // [128][64]
#define OW_FW2 14336    // [128][128]
#define OW_FW3 30720    // [128][128]
#define OW_RW1 47104    // [64][160]            (rows 155 -> 160)
#define OW_RW2 57344    // [32][64]
#define OW_TOTAL 59392  // halfs

__device__ __forceinline__ void lds_fence() {
    asm volatile("s_wait_dscnt 0" ::: "memory");
}

// Async global->LDS copy of 16 bytes (CDNA5 ASYNCcnt path).
__device__ __forceinline__ void async_copy16(void* ldsdst, const void* gsrc) {
    unsigned lo = (unsigned)(uintptr_t)ldsdst;   // low 32 bits = group-segment offset
    asm volatile("global_load_async_to_lds_b128 %0, %1, off"
                 :: "v"(lo), "v"(gsrc) : "memory");
}
__device__ __forceinline__ void wait_async() {
    asm volatile("s_wait_asynccnt 0x0" ::: "memory");
}

// A fragment (16x32 f16, M-rows in LDS, row r = lane%16, g = lane/16):
// element j: K = ko + (j<8 ? g*8+j : 16+g*8+(j-8))  -> two contiguous 16B chunks
__device__ __forceinline__ v16h frag_act(const _Float16* base, int stride, int r, int ko, int g) {
    AF a;
    const _Float16* p = base + r * stride + ko + g * 8;
    a.h[0] = *reinterpret_cast<const v8h*>(p);
    a.h[1] = *reinterpret_cast<const v8h*>(p + 16);
    return a.v;
}

// B fragment (32x16 f16) from col-major weights Wt[col][Kpad]:
// element j: K = ko + (j<8 ? g*8+j : 16+g*8+(j-8))  -> two contiguous 16B chunks
__device__ __forceinline__ v16h frag_w(const _Float16* Wt, int Kpad, int col, int ko, int g) {
    AF b;
    const _Float16* p = Wt + col * Kpad + ko + g * 8;
    b.h[0] = *reinterpret_cast<const v8h*>(p);
    b.h[1] = *reinterpret_cast<const v8h*>(p + 16);
    return b.v;
}

// One wave computes a [16 rows x Ndim] layer: out = act(in @ W + b)
__device__ __forceinline__ void mlp_layer(const _Float16* in, int istr,
                                          _Float16* out, int ostr,
                                          const _Float16* Wt, int Kdim, int Ndim,
                                          const float* bias, bool relu, int lane) {
    const int cl = lane & 15, g = lane >> 4;
    for (int nb = 0; nb < Ndim; nb += 16) {
        float bv = bias[nb + cl];
        v8f acc;
#pragma unroll
        for (int i = 0; i < 8; ++i) acc[i] = bv;
        for (int ko = 0; ko < Kdim; ko += 32) {
            v16h a = frag_act(in, istr, cl, ko, g);
            v16h b = frag_w(Wt, Kdim, nb + cl, ko, g);
            acc = __builtin_amdgcn_wmma_f32_16x16x32_f16(false, a, false, b,
                                                         (short)0, acc, false, false);
        }
#pragma unroll
        for (int i = 0; i < 8; ++i) {
            float v = acc[i];
            if (relu) v = fmaxf(v, 0.f);
            out[(i + 8 * g) * ostr + nb + cl] = (_Float16)v;
        }
    }
    lds_fence();
}

// HarmonicEmbedding(n=4, logspace, append_input): [sin(e)12, cos(e)12, x 3]
__device__ __forceinline__ void harmonic27(const float* x, float* out) {
    const float fr[4] = {1.f, 2.f, 4.f, 8.f};
#pragma unroll
    for (int d = 0; d < 3; ++d) {
#pragma unroll
        for (int fi = 0; fi < 4; ++fi) {
            float e = x[d] * fr[fi];
            out[d * 4 + fi]      = sinf(e);
            out[12 + d * 4 + fi] = cosf(e);
        }
        out[24 + d] = x[d];
    }
}

// ---------------- prep: f16 transposed weights + packed (x,y,z,|p|^2) table ----------
__device__ __forceinline__ void conv_t(const float* W, _Float16* Wt,
                                       int rows, int cols, int Kpad, int tid, int nt) {
    for (int i = tid; i < cols * Kpad; i += nt) {
        int c = i / Kpad, r = i % Kpad;
        Wt[i] = (r < rows) ? (_Float16)W[r * cols + c] : (_Float16)0.f;
    }
}

__global__ void prep_kernel(const float* pw1, const float* pw2, const float* fw1,
                            const float* fw2, const float* fw3, const float* rw1,
                            const float* rw2, const float* pts,
                            _Float16* wb, float4* pts4) {
    int tid = blockIdx.x * blockDim.x + threadIdx.x;
    int nt = gridDim.x * blockDim.x;
    conv_t(pw1, wb + OW_PW1,  59,  64,  64, tid, nt);
    conv_t(pw2, wb + OW_PW2,  64,  32,  64, tid, nt);
    conv_t(fw1, wb + OW_FW1,  64, 128,  64, tid, nt);
    conv_t(fw2, wb + OW_FW2, 128, 128, 128, tid, nt);
    conv_t(fw3, wb + OW_FW3, 128, 128, 128, tid, nt);
    conv_t(rw1, wb + OW_RW1, 155,  64, 160, tid, nt);
    conv_t(rw2, wb + OW_RW2,  64,  32,  64, tid, nt);
    for (int i = tid; i < NB * PP; i += nt) {
        float x = pts[i * 3 + 0], y = pts[i * 3 + 1], z = pts[i * 3 + 2];
        pts4[i] = make_float4(x, y, z, x * x + y * y + z * z);
    }
}

// ---------------- KNN: top-4 via double-buffered async LDS tiles ---------------------
#define TILE 1024
__global__ __launch_bounds__(256) void knn_kernel(const float* __restrict__ coords,
                                                  const float4* __restrict__ pts4,
                                                  float* __restrict__ dist,
                                                  int* __restrict__ idx) {
    __shared__ float4 tile[2][TILE];
    const int n = blockIdx.y;
    const int m = blockIdx.x * blockDim.x + threadIdx.x;
    const int nm = n * MM + m;
    const float cx = coords[nm * 3 + 0], cy = coords[nm * 3 + 1], cz = coords[nm * 3 + 2];
    const float cc  = cx * cx + cy * cy + cz * cz;
    const float m2x = -2.f * cx, m2y = -2.f * cy, m2z = -2.f * cz;
    const float4* src = pts4 + n * PP;

    float b0 = 3.4e38f, b1 = 3.4e38f, b2 = 3.4e38f, b3 = 3.4e38f;
    int i0 = 0, i1 = 0, i2 = 0, i3 = 0;

    const int ntile = (PP + TILE - 1) / TILE;
    // prologue: async-fill tile 0
    {
        int cnt = min(TILE, PP);
        for (int i = threadIdx.x; i < cnt; i += blockDim.x)
            async_copy16(&tile[0][i], src + i);
        wait_async();
        __syncthreads();
    }
    for (int t = 0; t < ntile; ++t) {
        const int t0 = t * TILE;
        const int cnt = min(TILE, PP - t0);
        if (t + 1 < ntile) {                    // overlap next tile's copy
            int c2 = min(TILE, PP - (t0 + TILE));
            for (int i = threadIdx.x; i < c2; i += blockDim.x)
                async_copy16(&tile[(t + 1) & 1][i], src + t0 + TILE + i);
        }
        const float4* buf = tile[t & 1];
        for (int j = 0; j < cnt; ++j) {
            float4 p = buf[j];
            float d2 = fmaf(m2x, p.x, fmaf(m2y, p.y, fmaf(m2z, p.z, cc + p.w)));
            if (d2 < b3) {
                b3 = d2; i3 = t0 + j;
                if (b3 < b2) {
                    float tt = b2; b2 = b3; b3 = tt; int ti = i2; i2 = i3; i3 = ti;
                    if (b2 < b1) {
                        tt = b1; b1 = b2; b2 = tt; ti = i1; i1 = i2; i2 = ti;
                        if (b1 < b0) { tt = b0; b0 = b1; b1 = tt; ti = i0; i0 = i1; i1 = ti; }
                    }
                }
            }
        }
        wait_async();
        __syncthreads();
    }
    const int base = nm * KK;
    dist[base + 0] = b0; dist[base + 1] = b1; dist[base + 2] = b2; dist[base + 3] = b3;
    idx[base + 0] = i0;  idx[base + 1] = i1;  idx[base + 2] = i2;  idx[base + 3] = i3;
}

// ---------------- wsum[n,k] = sum_m 1/dist[n,m,k] ------------------------------------
__global__ __launch_bounds__(256) void wsum_kernel(const float* __restrict__ dist,
                                                   float* __restrict__ wsum) {
    __shared__ float red[256];
    const int nk = blockIdx.x, n = nk / KK, k = nk % KK;
    float s = 0.f;
    for (int m = threadIdx.x; m < MM; m += blockDim.x)
        s += 1.f / dist[(n * MM + m) * KK + k];
    red[threadIdx.x] = s;
    __syncthreads();
    for (int off = 128; off > 0; off >>= 1) {
        if ((int)threadIdx.x < off) red[threadIdx.x] += red[threadIdx.x + off];
        __syncthreads();
    }
    if (threadIdx.x == 0) wsum[nk] = red[0];
}

// ---------------- triplane bilinear sampling, mean over 3 planes ---------------------
__device__ __forceinline__ float tap(const float* img, float fx, float fy) {
    bool v = (fx >= 0.f) & (fx < (float)HPD) & (fy >= 0.f) & (fy < (float)HPD);
    int ix = min(max((int)fx, 0), HPD - 1);
    int iy = min(max((int)fy, 0), HPD - 1);
    return v ? img[iy * HPD + ix] : 0.f;
}

__global__ __launch_bounds__(256) void triplane_kernel(const float* __restrict__ coords,
                                                       const float* __restrict__ tp,
                                                       float* __restrict__ tex) {
    const int c = threadIdx.x & 31;                       // channel = lane
    const int nm = blockIdx.x * 8 + (threadIdx.x >> 5);   // combined n*M+m
    const int n = nm / MM;
    const float c0 = coords[nm * 3 + 0], c1 = coords[nm * 3 + 1], c2 = coords[nm * 3 + 2];
    float acc = 0.f;
#pragma unroll
    for (int pl = 0; pl < 3; ++pl) {
        float gx = (pl == 2) ? c2 : c0;
        float gy = (pl == 0) ? c1 : ((pl == 1) ? c2 : c0);
        const float* img = tp + (size_t)((n * 3 + pl) * CPL + c) * (HPD * HPD);
        float x = (gx + 1.f) * (HPD * 0.5f) - 0.5f;
        float y = (gy + 1.f) * (HPD * 0.5f) - 0.5f;
        float x0 = floorf(x), y0 = floorf(y);
        float wx1 = x - x0, wx0 = 1.f - wx1;
        float wy1 = y - y0, wy0 = 1.f - wy1;
        acc += tap(img, x0,       y0      ) * wx0 * wy0
             + tap(img, x0 + 1.f, y0      ) * wx1 * wy0
             + tap(img, x0,       y0 + 1.f) * wx0 * wy1
             + tap(img, x0 + 1.f, y0 + 1.f) * wx1 * wy1;
    }
    tex[nm * 32 + c] = acc * (1.f / 3.f);
}

// ---------------- per-neighbor MLP (WMMA) + weighted K-sum ---------------------------
__global__ __launch_bounds__(256) void point_mlp_kernel(
        const float* __restrict__ coords, const float* __restrict__ pts,
        const float* __restrict__ emb, const int* __restrict__ idx,
        const float* __restrict__ dist, const float* __restrict__ wsum,
        const _Float16* __restrict__ wb, const float* __restrict__ pb1,
        const float* __restrict__ pb2, float* __restrict__ ptsfeat) {
    __shared__ __align__(32) _Float16 sm[8 * 2048];
    const int lane = threadIdx.x & 31, wave = threadIdx.x >> 5;
    _Float16* act = sm + wave * 2048;     // [16][64]
    _Float16* h1  = act + 1024;           // [16][64]
    const int rowbase = (blockIdx.x * 8 + wave) * 16;   // rows = (n*M+m)*K + k
    const int r = lane & 15, g = lane >> 4;
    const int row = rowbase + r;
    const int pidx = idx[row];
    if (g == 0) {
#pragma unroll
        for (int j = 0; j < 32; ++j)
            act[r * 64 + j] = (_Float16)emb[pidx * 32 + j];
    } else {
        const int nm = row >> 2;          // /K
        const int n  = row >> 16;         // /(M*K)
        float cx = coords[nm * 3 + 0], cy = coords[nm * 3 + 1], cz = coords[nm * 3 + 2];
        const float* pp = pts + (size_t)(n * PP + pidx) * 3;
        float rx = cx - pp[0], ry = cy - pp[1], rz = cz - pp[2];
        float nrm = fmaxf(sqrtf(rx * rx + ry * ry + rz * rz), 1e-12f);
        float rel[3] = {rx / nrm, ry / nrm, rz / nrm};
        float h[27];
        harmonic27(rel, h);
#pragma unroll
        for (int j = 0; j < 27; ++j) act[r * 64 + 32 + j] = (_Float16)h[j];
#pragma unroll
        for (int j = 27; j < 32; ++j) act[r * 64 + 32 + j] = (_Float16)0.f;
    }
    lds_fence();
    mlp_layer(act, 64, h1, 64, wb + OW_PW1, 64, 64, pb1, true, lane);

    // Layer 2 (64->32) with fused inverse-distance weighted sum over k (rows 4m+k).
    float w8[8];
#pragma unroll
    for (int i = 0; i < 8; ++i) {
        int rr = rowbase + 8 * g + i;
        int k = rr & 3, n2 = rr >> 16;
        w8[i] = (1.f / dist[rr]) / wsum[n2 * KK + k];
    }
    const int cl = lane & 15;
    const int mb = rowbase >> 2;          // combined n*M+m base for this wave (4 m's)
#pragma unroll
    for (int nb = 0; nb < 32; nb += 16) {
        float bv = pb2[nb + cl];
        v8f acc;
#pragma unroll
        for (int i = 0; i < 8; ++i) acc[i] = bv;
#pragma unroll
        for (int ko = 0; ko < 64; ko += 32) {
            v16h a = frag_act(h1, 64, cl, ko, g);
            v16h b = frag_w(wb + OW_PW2, 64, nb + cl, ko, g);
            acc = __builtin_amdgcn_wmma_f32_16x16x32_f16(false, a, false, b,
                                                         (short)0, acc, false, false);
        }
        float s0 = 0.f, s1 = 0.f;
#pragma unroll
        for (int i = 0; i < 4; ++i) s0 += acc[i] * w8[i];
#pragma unroll
        for (int i = 4; i < 8; ++i) s1 += acc[i] * w8[i];
        ptsfeat[(mb + 2 * g + 0) * 32 + nb + cl] = s0;
        ptsfeat[(mb + 2 * g + 1) * 32 + nb + cl] = s1;
    }
}

// ---------------- dense MLP stack + heads (WMMA) -------------------------------------
__global__ __launch_bounds__(128) void main_mlp_kernel(
        const float* __restrict__ coords, const float* __restrict__ dirs,
        const float* __restrict__ tex, const float* __restrict__ ptsfeat,
        const _Float16* __restrict__ wb,
        const float* __restrict__ fb1, const float* __restrict__ fb2,
        const float* __restrict__ fb3, const float* __restrict__ dw,
        const float* __restrict__ db, const float* __restrict__ rb1,
        const float* __restrict__ rb2,
        float* __restrict__ dens, float* __restrict__ rgb) {
    __shared__ __align__(32) _Float16 sm[4 * 4608];
    const int lane = threadIdx.x & 31, wave = threadIdx.x >> 5;
    _Float16* bufA = sm + wave * 4608;    // [16][128]
    _Float16* bufB = bufA + 2048;         // [16][160]
    const int rowbase = (blockIdx.x * 4 + wave) * 16;   // rows = n*M+m
    const int r = lane & 15, g = lane >> 4;
    const int row = rowbase + r;

    // stage feat = [tex(32) | ptsfeat(32)] as f16
    if (g == 0) {
#pragma unroll
        for (int j = 0; j < 32; ++j) bufA[r * 128 + j] = (_Float16)tex[row * 32 + j];
    } else {
#pragma unroll
        for (int j = 0; j < 32; ++j) bufA[r * 128 + 32 + j] = (_Float16)ptsfeat[row * 32 + j];
    }
    lds_fence();

    mlp_layer(bufA, 128, bufB, 160, wb + OW_FW1,  64, 128, fb1, true,  lane);
    mlp_layer(bufB, 160, bufA, 128, wb + OW_FW2, 128, 128, fb2, true,  lane);
    mlp_layer(bufA, 128, bufB, 160, wb + OW_FW3, 128, 128, fb3, false, lane);  // feat

    // density head: dot(feat,dw)+db, softplus(10x)/10 * selector, 1-exp(-raw)
    float part = 0.f;
#pragma unroll
    for (int j = 0; j < 64; ++j)
        part += (float)bufB[r * 160 + g * 64 + j] * dw[g * 64 + j];
    part += __shfl_xor(part, 16, 32);
    if (g == 0) {
        float t = 10.f * (part + db[0]);
        float sp = (t > 20.f) ? t : log1pf(expf(t));
        float raw = sp * 0.1f;
        float a0 = coords[row * 3 + 0], a1 = coords[row * 3 + 1], a2 = coords[row * 3 + 2];
        bool sel = (a0 > -1.f) & (a0 < 1.f) & (a1 > -1.f) & (a1 < 1.f)
                 & (a2 > -1.f) & (a2 < 1.f);
        if (!sel) raw = 0.f;
        dens[row] = 1.f - expf(-raw);
    }

    // harmonic(normalized rays) into bufB cols 128..159 (pad zero)
    if (g == 0) {
        float dx = dirs[row * 3 + 0], dy = dirs[row * 3 + 1], dz = dirs[row * 3 + 2];
        float nrm = fmaxf(sqrtf(dx * dx + dy * dy + dz * dz), 1e-12f);
        float rr3[3] = {dx / nrm, dy / nrm, dz / nrm};
        float h[27];
        harmonic27(rr3, h);
#pragma unroll
        for (int j = 0; j < 27; ++j) bufB[r * 160 + 128 + j] = (_Float16)h[j];
#pragma unroll
        for (int j = 27; j < 32; ++j) bufB[r * 160 + 128 + j] = (_Float16)0.f;
    }
    lds_fence();

    mlp_layer(bufB, 160, bufA, 128, wb + OW_RW1, 160, 64, rb1, true, lane);

    // final 64->32 with sigmoid epilogue on first 3 channels
    const int cl = r;
#pragma unroll
    for (int nb = 0; nb < 32; nb += 16) {
        float bv = rb2[nb + cl];
        v8f acc;
#pragma unroll
        for (int i = 0; i < 8; ++i) acc[i] = bv;
#pragma unroll
        for (int ko = 0; ko < 64; ko += 32) {
            v16h a = frag_act(bufA, 128, cl, ko, g);
            v16h b = frag_w(wb + OW_RW2, 64, nb + cl, ko, g);
            acc = __builtin_amdgcn_wmma_f32_16x16x32_f16(false, a, false, b,
                                                         (short)0, acc, false, false);
        }
#pragma unroll
        for (int i = 0; i < 8; ++i) {
            int rg = rowbase + i + 8 * g;
            int cg = nb + cl;
            float v = acc[i];
            if (cg < 3) v = 1.002f / (1.f + expf(-v)) - 0.001f;
            rgb[(size_t)rg * 32 + cg] = v;
        }
    }
}

// -------------------------------------------------------------------------------------
extern "C" void kernel_launch(void* const* d_in, const int* in_sizes, int n_in,
                              void* d_out, int out_size, void* d_ws, size_t ws_size,
                              hipStream_t stream) {
    const float* coords = (const float*)d_in[0];
    const float* dirs   = (const float*)d_in[1];
    const float* pts    = (const float*)d_in[2];
    const float* tp     = (const float*)d_in[3];
    const float* emb    = (const float*)d_in[4];
    const float* pw1 = (const float*)d_in[5];  const float* pb1 = (const float*)d_in[6];
    const float* pw2 = (const float*)d_in[7];  const float* pb2 = (const float*)d_in[8];
    const float* fw1 = (const float*)d_in[9];  const float* fb1 = (const float*)d_in[10];
    const float* fw2 = (const float*)d_in[11]; const float* fb2 = (const float*)d_in[12];
    const float* fw3 = (const float*)d_in[13]; const float* fb3 = (const float*)d_in[14];
    const float* dw  = (const float*)d_in[15]; const float* db  = (const float*)d_in[16];
    const float* rw1 = (const float*)d_in[17]; const float* rb1 = (const float*)d_in[18];
    const float* rw2 = (const float*)d_in[19]; const float* rb2 = (const float*)d_in[20];

    float* out  = (float*)d_out;
    float* dens = out;                              // [N,M,1]
    float* rgb  = out + NB * MM;                    // [N,M,32]
    float* dist = out + NB * MM + NB * MM * 32;     // [N,M,K]

    char* ws = (char*)d_ws;
    int*      idx     = (int*)ws;                      // 131072 ints
    float*    wsum    = (float*)(ws + 524288);         // 8 floats
    float*    tex     = (float*)(ws + 524352);         // 4 MB
    float*    ptsfeat = (float*)(ws + 4718656);        // 4 MB
    _Float16* wbase   = (_Float16*)(ws + 8912960);     // ~116 KB f16 weights
    float4*   pts4    = (float4*)(ws + 9031744);       // packed (x,y,z,|p|^2)

    prep_kernel<<<64, 256, 0, stream>>>(pw1, pw2, fw1, fw2, fw3, rw1, rw2, pts,
                                        wbase, pts4);
    knn_kernel<<<dim3(MM / 256, NB), 256, 0, stream>>>(coords, pts4, dist, idx);
    wsum_kernel<<<NB * KK, 256, 0, stream>>>(dist, wsum);
    triplane_kernel<<<(NB * MM) / 8, 256, 0, stream>>>(coords, tp, tex);
    point_mlp_kernel<<<(NB * MM * KK) / 128, 256, 0, stream>>>(
        coords, pts, emb, idx, dist, wsum, wbase, pb1, pb2, ptsfeat);
    main_mlp_kernel<<<(NB * MM) / 64, 128, 0, stream>>>(
        coords, dirs, tex, ptsfeat, wbase, fb1, fb2, fb3, dw, db, rb1, rb2, dens, rgb);
}
